// GAT_45810121179415
// MI455X (gfx1250) — matrix operand order repaired and checked by reading, not compiled, not run on hardware
//
#include <hip/hip_runtime.h>
#include <hip/hip_bf16.h>

typedef __attribute__((ext_vector_type(16))) __bf16 v16bf;
typedef __attribute__((ext_vector_type(8)))  __bf16 v8bf;
typedef __attribute__((ext_vector_type(8)))  float  v8f;

#define XROW 264   // padded row stride (bf16 elems) for x tile in LDS
#define WROW 264   // padded row stride (f32 elems) for wx tile in LDS

__device__ __forceinline__ float gelu_exact(float x) {
    // exact erf-based gelu, matches jax.nn.gelu(approximate=False)
    return 0.5f * x * (1.0f + erff(x * 0.70710678118654752f));
}

union ABfrag { v16bf v; v8bf h[2]; };

// G groups of (16 nodes x 256 features) per workgroup; B-fragments (weights)
// are reused across the G groups -> GxL2-traffic reduction on weights.
//   xin : [Gtot, 16, 256] f32
//   Wt  : [4, 256, 256] bf16, n-major (Wt[h][n][k] = W[h][k][n])
//   Avec: [4, 512] f32
//   out : [Gtot, 256] f32  (mean over 4 heads of gelu-attention reduction)
template<int G>
__global__ __launch_bounds__(256, 2) void gat_layer_kernel(
    const float* __restrict__ xin,
    const __bf16* __restrict__ Wt,
    const float* __restrict__ Avec,
    float* __restrict__ out)
{
    // Hand-placed LDS arena: xA at offset 0 so every GEMM-loop ds_load_b128
    // fits a 16-bit immediate offset (no per-iteration address VALU and no
    // WMMA->VALU hazard v_nops in the hot loop). wx/zbuf live above it and
    // are only touched in the barrier-separated epilogue.
    __shared__ __align__(16) unsigned char smem[
        G * 16 * XROW * 2 + G * 16 * WROW * 4 + G * 16 * 4];
    __bf16* xA   = (__bf16*)smem;                              // [G][16*XROW]
    float*  wx   = (float*)(smem + G * 16 * XROW * 2);         // [G][16*WROW]
    float*  zbuf = (float*)(smem + G * 16 * XROW * 2
                                 + G * 16 * WROW * 4);         // [G][16]

    const int g0   = blockIdx.x * G;
    const int t    = threadIdx.x;      // 0..255
    const int lane = t & 31;
    const int wv   = t >> 5;           // wave 0..7
    const int nn   = lane & 15;        // M (A-frag) / N (B,D-frag) index
    const int half = lane >> 4;        // lane half selects K-range / row block

    // ---- stage G x-tiles into LDS as bf16 (coalesced f32 reads) ----
    #pragma unroll
    for (int r = 0; r < 16 * G; ++r) {
        const int gg = r >> 4, rr = r & 15;
        xA[gg * (16 * XROW) + rr * XROW + t] =
            (__bf16)xin[((size_t)(g0 + gg) * 16 + rr) * 256 + t];
    }
    __syncthreads();

    const int chunk = t & 15;          // 16-wide dot-product partial index
    const int nrow  = t >> 4;          // node row 0..15 for score stage

    // per-lane A-fragment base: constant offsets g*8448 + kc*64 (+16) on top
    const __bf16* xAbase = xA + nn * XROW + half * 16;

    float headsum[G];
    #pragma unroll
    for (int g = 0; g < G; ++g) headsum[g] = 0.0f;

    for (int h = 0; h < 4; ++h) {
        // ====== GEMM: wx[g] = xA[g] @ W[h]  (bf16 WMMA, f32 acc) ============
        const __bf16* Wh = Wt + h * 65536;
        v8f acc0[G], acc1[G];
        #pragma unroll
        for (int g = 0; g < G; ++g) { acc0[g] = {}; acc1[g] = {}; }
        const int n0 = wv * 16;          // this wave's two output column tiles
        const int n1 = (wv + 8) * 16;
        // unroll 2: enough to double-buffer loads vs WMMA while keeping the
        // live register set < 256 VGPRs (no s_set_vgpr_msb, >=2 waves/SIMD)
        #pragma unroll 2
        for (int kc = 0; kc < 8; ++kc) {
            // B fragments: lane owns a column; 16 contiguous k values
            const int kb = kc * 32 + half * 16;
            const v16bf b0 = *(const v16bf*)&Wh[(n0 + nn) * 256 + kb];
            const v16bf b1 = *(const v16bf*)&Wh[(n1 + nn) * 256 + kb];
            #pragma unroll
            for (int g = 0; g < G; ++g) {
                // A fragment: 16 contiguous k per lane half (k = 16*half + s), same order as B
                const __bf16* ap = xAbase + g * (16 * XROW) + kc * 32;
                ABfrag a;
                a.h[0] = *(const v8bf*)(ap);
                a.h[1] = *(const v8bf*)(ap + 8);
                acc0[g] = __builtin_amdgcn_wmma_f32_16x16x32_bf16(
                    false, a.v, false, b0, (short)0, acc0[g], false, false);
                acc1[g] = __builtin_amdgcn_wmma_f32_16x16x32_bf16(
                    false, a.v, false, b1, (short)0, acc1[g], false, false);
            }
        }
        // D layout: lane -> col nn, VGPR v -> row v + half*8
        #pragma unroll
        for (int g = 0; g < G; ++g) {
            #pragma unroll
            for (int v = 0; v < 8; ++v) {
                wx[g * (16 * WROW) + (v + half * 8) * WROW + n0 + nn] = acc0[g][v];
                wx[g * (16 * WROW) + (v + half * 8) * WROW + n1 + nn] = acc1[g][v];
            }
        }
        __syncthreads();

        // ====== attention scores =============================================
        // raw[n] = wx[n,:] . A[h][:256] + wx[0,:] . A[h][256:512]
        const float* Ah = Avec + h * 512;
        #pragma unroll
        for (int g = 0; g < G; ++g) {
            const float* wg = wx + g * (16 * WROW);
            float p = 0.0f, p2 = 0.0f;
            #pragma unroll
            for (int i = 0; i < 16; ++i) {
                const int c = chunk * 16 + i;
                p  += wg[nrow * WROW + c] * Ah[c];
                p2 += wg[c] * Ah[256 + c];          // root row (n = 0)
            }
            #pragma unroll
            for (int o = 8; o >= 1; o >>= 1) {
                p  += __shfl_xor(p,  o, 16);
                p2 += __shfl_xor(p2, o, 16);
            }
            if (chunk == 0)
                zbuf[g * 16 + nrow] = gelu_exact(gelu_exact(p + p2));
        }
        __syncthreads();

        // ====== softmax over n=16, weighted reduce, gelu =====================
        #pragma unroll
        for (int g = 0; g < G; ++g) {
            const float* wg = wx + g * (16 * WROW);
            const float* zg = zbuf + g * 16;
            float mx = zg[0];
            #pragma unroll
            for (int n = 1; n < 16; ++n) mx = fmaxf(mx, zg[n]);
            float den = 0.0f;
            #pragma unroll
            for (int n = 0; n < 16; ++n) den += __expf(zg[n] - mx);
            const float inv = 1.0f / den;
            float acc = 0.0f;
            #pragma unroll
            for (int n = 0; n < 16; ++n)
                acc += wg[n * WROW + t] * (__expf(zg[n] - mx) * inv);
            headsum[g] += gelu_exact(acc);
        }
        __syncthreads();    // wx/zbuf reused by next head
    }

    #pragma unroll
    for (int g = 0; g < G; ++g)
        out[(size_t)(g0 + g) * 256 + t] = 0.25f * headsum[g];  // head mean
}

// Convert f32 weights [4,256,256] (k-major) -> bf16 transposed [4,256,256] (n-major).
__global__ __launch_bounds__(256) void convert_wt_kernel(
    const float* __restrict__ W, __bf16* __restrict__ Wt)
{
    const int id = blockIdx.x * 256 + threadIdx.x;   // < 4*65536
    const int h = id >> 16;
    const int r = id & 65535;
    const int n = r >> 8;
    const int k = r & 255;
    Wt[id] = (__bf16)W[h * 65536 + k * 256 + n];
}

extern "C" void kernel_launch(void* const* d_in, const int* in_sizes, int n_in,
                              void* d_out, int out_size, void* d_ws, size_t ws_size,
                              hipStream_t stream) {
    const float* x  = (const float*)d_in[0];  // [512,16,16,256]
    const float* W0 = (const float*)d_in[1];  // [4,256,256]
    const float* A0 = (const float*)d_in[2];  // [4,512,1]
    const float* W1 = (const float*)d_in[3];  // [4,256,256]
    const float* A1 = (const float*)d_in[4];  // [4,512,1]
    float* out = (float*)d_out;               // [512,256]

    // workspace layout: W0t (512KB) | W1t (512KB) | x1 f32 (8MB)
    __bf16* W0t = (__bf16*)d_ws;
    __bf16* W1t = W0t + 4 * 65536;
    float*  x1  = (float*)((char*)d_ws + (1 << 20));

    convert_wt_kernel<<<1024, 256, 0, stream>>>(W0, W0t);
    convert_wt_kernel<<<1024, 256, 0, stream>>>(W1, W1t);

    // layer 1: 8192 groups, 4 per block -> 2048 blocks
    gat_layer_kernel<4><<<2048, 256, 0, stream>>>(x, W0t, A0, x1);
    // layer 2: 512 groups, 2 per block -> 256 blocks
    gat_layer_kernel<2><<<256, 256, 0, stream>>>(x1, W1t, A1, out);
}
